// RNNMod_7241314861316
// MI455X (gfx1250) — compile-verified, matrix operand order checked by reading, not run on hardware
//
#include <hip/hip_runtime.h>

#define BATCH 4096
#define TT    140
#define HH    128      // hidden
#define OSZ   5
#define BM    16       // batch rows per workgroup

typedef __attribute__((ext_vector_type(16))) _Float16 v16h;
typedef __attribute__((ext_vector_type(8)))  float    v8f;
typedef __attribute__((ext_vector_type(4)))  float    v4f;

union F16Frag {
    v16h     v;
    v4f      f4[2];
    _Float16 h[16];
};

// sigmoid(x) = 1/(1+e^-x): saturates correctly without clamps
// (exp -> inf => rcp -> 0; exp -> 0 => 1)
__device__ __forceinline__ float fsigmoid(float x) {
    return __builtin_amdgcn_rcpf(1.f + __expf(-x));
}
// tanh(x) = 1 - 2/(e^{2x}+1): saturates to +/-1, no inf/inf NaN path
__device__ __forceinline__ float ftanh_(float x) {
    return 1.f - 2.f * __builtin_amdgcn_rcpf(1.f + __expf(2.f * x));
}

// Build a WMMA B-fragment (16x16x32_f16 layout) for one lane from a row-major
// f32 weight row: elems 0..7 = W[row][k0+off+e], elems 8..15 = W[row][k0+16+off+e]
__device__ __forceinline__ F16Frag make_bfrag(const float* __restrict__ wrow,
                                              int k0, int off) {
    const v4f lo0 = *(const v4f*)(wrow + k0 + off);
    const v4f lo1 = *(const v4f*)(wrow + k0 + off + 4);
    const v4f hi0 = *(const v4f*)(wrow + k0 + 16 + off);
    const v4f hi1 = *(const v4f*)(wrow + k0 + 16 + off + 4);
    F16Frag fr;
#pragma unroll
    for (int e = 0; e < 4; ++e) {
        fr.h[e]      = (_Float16)lo0[e];
        fr.h[4 + e]  = (_Float16)lo1[e];
        fr.h[8 + e]  = (_Float16)hi0[e];
        fr.h[12 + e] = (_Float16)hi1[e];
    }
    return fr;
}

extern "C" __global__ __launch_bounds__(256, 1)
void lstm_wmma_kernel(const float* __restrict__ x,
                      const float* __restrict__ W_ih,
                      const float* __restrict__ W_hh,
                      const float* __restrict__ b_ih,
                      const float* __restrict__ b_hh,
                      const float* __restrict__ W1,
                      const float* __restrict__ b1,
                      const float* __restrict__ W2,
                      const float* __restrict__ b2,
                      float* __restrict__ out)
{
    // ~25 KB static LDS: double-buffered h tile + x tile + r1
    __shared__ __align__(32) _Float16 hbuf[2][BM * HH];  // 2 * 4 KB
    __shared__ __align__(16) float    xbuf[BM * TT];     // 8.75 KB
    __shared__ __align__(16) float    r1buf[BM * HH];    // 8 KB

    const int tid    = threadIdx.x;
    const int lane   = tid & 31;
    const int wave   = tid >> 5;                 // 0..7
    const int batch0 = blockIdx.x * BM;

    const int col  = lane & 15;                  // N within this wave's 16-wide tile
    const int roff = (lane >> 4) * 8;            // C-fragment row offset (0 or 8)
    const int hcol = wave * 16 + col;            // hidden column owned by this wave
    const int off  = (lane < 16) ? 0 : 8;        // f16 A/B fragment half-select

    // ---- W_hh B-fragments: resident in VGPRs for the whole sequence ----
    // gates[m][n] = sum_k h[m][k]*W_hh[n][k]; B[k][n] = W_hh[n][k];
    // lane's column n = g*128 + hcol -> one W_hh row per gate per lane.
    F16Frag bw[4][4];                            // [gate][ktile] = 128 VGPRs
#pragma unroll
    for (int g = 0; g < 4; ++g) {
        const float* wrow = W_hh + (size_t)(g * HH + hcol) * HH;
#pragma unroll
        for (int kt = 0; kt < 4; ++kt)
            bw[g][kt] = make_bfrag(wrow, kt * 32, off);
    }

    // ---- per-lane gate constants ----
    float wig[4], bg[4];
#pragma unroll
    for (int g = 0; g < 4; ++g) {
        const int n = g * HH + hcol;
        wig[g] = W_ih[n];
        bg[g]  = b_ih[n] + b_hh[n];
    }

    // ---- x tile into LDS; h buffer 0 = 0 ----
    for (int i = tid; i < BM * TT; i += 256) {
        const int m = i & 15;
        const int t = i >> 4;
        xbuf[i] = x[(size_t)(batch0 + m) * TT + t];
    }
    for (int i = tid; i < BM * HH; i += 256) hbuf[0][i] = (_Float16)0.f;

    float cst[8];
#pragma unroll
    for (int r = 0; r < 8; ++r) cst[r] = 0.f;

    __syncthreads();

    const int am = lane & 15;                    // A-fragment row = lane%16

    // ================= recurrent loop (1 barrier/step) =================
    for (int t = 0; t < TT; ++t) {
        const _Float16* hr = hbuf[t & 1];
        _Float16*       hw = hbuf[(t + 1) & 1];

        // A fragments from shared h tile (per §7.12.2 16-bit A layout)
        F16Frag a[4];
#pragma unroll
        for (int kt = 0; kt < 4; ++kt) {
            const int k0 = kt * 32;
            a[kt].f4[0] = *(const v4f*)(hr + am * HH + k0 + off);
            a[kt].f4[1] = *(const v4f*)(hr + am * HH + k0 + 16 + off);
        }
        const v4f x0 = *(const v4f*)(xbuf + t * 16 + roff);
        const v4f x1 = *(const v4f*)(xbuf + t * 16 + roff + 4);
        float xv[8] = {x0[0], x0[1], x0[2], x0[3], x1[0], x1[1], x1[2], x1[3]};

        v8f acc[4];
#pragma unroll
        for (int g = 0; g < 4; ++g) {
#pragma unroll
            for (int r = 0; r < 8; ++r) acc[g][r] = bg[g] + xv[r] * wig[g];
        }
        // 16 back-to-back WMMAs, all operands in registers
#pragma unroll
        for (int g = 0; g < 4; ++g) {
#pragma unroll
            for (int kt = 0; kt < 4; ++kt) {
                acc[g] = __builtin_amdgcn_wmma_f32_16x16x32_f16(
                    false, a[kt].v, false, bw[g][kt].v, (short)0, acc[g],
                    false, false);
            }
        }

        // elementwise LSTM cell; c stays in registers
#pragma unroll
        for (int r = 0; r < 8; ++r) {
            const float iv = fsigmoid(acc[0][r]);
            const float fv = fsigmoid(acc[1][r]);
            const float gv = ftanh_(acc[2][r]);
            const float ov = fsigmoid(acc[3][r]);
            cst[r] = fv * cst[r] + iv * gv;
            const float hv = ov * ftanh_(cst[r]);
            hw[(roff + r) * HH + hcol] = (_Float16)hv;
        }
        __syncthreads();   // orders writes to next buffer vs reads of it
    }

    // ================= head: r1 = relu(h @ W1^T + b1) =================
    {
        const _Float16* hf = hbuf[TT & 1];       // TT even -> buffer 0
        F16Frag a[4];
#pragma unroll
        for (int kt = 0; kt < 4; ++kt) {
            const int k0 = kt * 32;
            a[kt].f4[0] = *(const v4f*)(hf + am * HH + k0 + off);
            a[kt].f4[1] = *(const v4f*)(hf + am * HH + k0 + 16 + off);
        }
        const float bb = b1[hcol];
        v8f c0;
#pragma unroll
        for (int r = 0; r < 8; ++r) c0[r] = bb;
        const float* w1row = W1 + (size_t)hcol * HH;
#pragma unroll
        for (int kt = 0; kt < 4; ++kt) {
            const F16Frag bf = make_bfrag(w1row, kt * 32, off);
            c0 = __builtin_amdgcn_wmma_f32_16x16x32_f16(
                    false, a[kt].v, false, bf.v, (short)0, c0, false, false);
        }
#pragma unroll
        for (int r = 0; r < 8; ++r)
            r1buf[(roff + r) * HH + hcol] = fmaxf(c0[r], 0.f);
    }
    __syncthreads();

    // ================= r2 = r1 @ W2^T + b2; log_softmax =================
    if (tid < BM) {
        const int m = tid;
        float z[OSZ];
#pragma unroll
        for (int o = 0; o < OSZ; ++o) {
            float s = b2[o];
            const float* wr = W2 + o * HH;
            for (int j = 0; j < HH; ++j) s += r1buf[m * HH + j] * wr[j];
            z[o] = s;
        }
        float mx = z[0];
#pragma unroll
        for (int o = 1; o < OSZ; ++o) mx = fmaxf(mx, z[o]);
        float se = 0.f;
#pragma unroll
        for (int o = 0; o < OSZ; ++o) se += __expf(z[o] - mx);
        const float lse = __logf(se);
#pragma unroll
        for (int o = 0; o < OSZ; ++o)
            out[(size_t)(batch0 + m) * OSZ + o] = z[o] - mx - lse;
    }
}

extern "C" void kernel_launch(void* const* d_in, const int* in_sizes, int n_in,
                              void* d_out, int out_size, void* d_ws, size_t ws_size,
                              hipStream_t stream) {
    const float* x    = (const float*)d_in[0];
    const float* W_ih = (const float*)d_in[1];
    const float* W_hh = (const float*)d_in[2];
    const float* b_ih = (const float*)d_in[3];
    const float* b_hh = (const float*)d_in[4];
    const float* W1   = (const float*)d_in[5];
    const float* b1   = (const float*)d_in[6];
    const float* W2   = (const float*)d_in[7];
    const float* b2   = (const float*)d_in[8];
    float* out = (float*)d_out;

    lstm_wmma_kernel<<<BATCH / BM, 256, 0, stream>>>(
        x, W_ih, W_hh, b_ih, b_hh, W1, b1, W2, b2, out);
}
